// WhatsnetLayer_85126251807354
// MI455X (gfx1250) — matrix operand order repaired and checked by reading, not compiled, not run on hardware
//
#include <hip/hip_runtime.h>
#include <hip/hip_bf16.h>

// WhatsnetLayer (2x SAB-ShawRE encoder + PrevQ MAB decoder + final Linear)
// One workgroup (128 threads = 4 wave32) per batch element; all intermediates
// LDS-resident in f16; GEMMs via v_wmma_f32_16x16x32_f16 with f32 accumulate.

typedef __attribute__((ext_vector_type(16))) _Float16 v16h;
typedef __attribute__((ext_vector_type(8)))  float    v8f;

#define LDH 136   // row stride (halves): 272B rows -> 16B-aligned fragments, bank-clean
#define LDA 72    // row stride (halves) for 64x64 attention matrix (144B rows)
#define LDT 72    // row stride (halves) for 128x64 transposed V (144B rows)

// Explicit global-address-space pointers: guarantees global_load_* (LOADcnt
// only) instead of flat_load_* (which ties LOADcnt *and* DScnt).
typedef const float __attribute__((address_space(1))) GCF;
typedef float       __attribute__((address_space(1))) GMF;
static __device__ __forceinline__ GCF* gptr(const float* p) {
  return (GCF*)(unsigned long long)(const void*)p;
}
static __device__ __forceinline__ GMF* gptr_mut(float* p) {
  return (GMF*)(unsigned long long)(void*)p;
}

// ---- WMMA fragment builders (wave32, contiguous k = kb+i in both A and B) ----------

// A operand: 16x32 f16 tile from row-major f16 LDS matrix. lane: M = lane%16,
// K = kb+i, kb = (lane>=16)?16:0.
static __device__ __forceinline__ v16h loadA16(const _Float16* S, int ld,
                                               int m0, int k0, int lane) {
  int row = m0 + (lane & 15);
  int kb  = (lane & 16) ? 16 : 0;
  const _Float16* p = S + row * ld + k0 + kb;
  v16h a;
#pragma unroll
  for (int i = 0; i < 8; ++i) a[i] = p[i];
#pragma unroll
  for (int i = 0; i < 8; ++i) a[8 + i] = p[8 + i];
  return a;
}

// B operand = W^T chunk (B[k][n] = W[n][k]), W row-major f32 in global (L2-hot).
// lane: N = lane%16, K = kb+i, kb = (lane>=16)?16:0.
static __device__ __forceinline__ v16h loadBT_f32(GCF* W, int ldw,
                                                  int n0, int k0, int lane) {
  int nn = n0 + (lane & 15);
  int kb = (lane & 16) ? 16 : 0;
  GCF* p = W + (size_t)nn * ldw + k0 + kb;
  v16h b;
#pragma unroll
  for (int i = 0; i < 16; ++i) b[i] = (_Float16)p[i];
  return b;
}

// B operand = M^T chunk from row-major f16 LDS matrix (B[k][n] = M[n][k]).
static __device__ __forceinline__ v16h loadBT_f16(const _Float16* S, int ld,
                                                  int n0, int k0, int lane) {
  int nn = n0 + (lane & 15);
  int kb = (lane & 16) ? 16 : 0;
  const _Float16* p = S + nn * ld + k0 + kb;
  v16h b;
#pragma unroll
  for (int i = 0; i < 16; ++i) b[i] = p[i];
  return b;
}

// ---- 16-lane / 32-lane shfl reductions -------------------------------------
static __device__ __forceinline__ float hmax16(float v) {
  v = fmaxf(v, __shfl_xor(v, 1, 32)); v = fmaxf(v, __shfl_xor(v, 2, 32));
  v = fmaxf(v, __shfl_xor(v, 4, 32)); v = fmaxf(v, __shfl_xor(v, 8, 32));
  return v;
}
static __device__ __forceinline__ float hsum16(float v) {
  v += __shfl_xor(v, 1, 32); v += __shfl_xor(v, 2, 32);
  v += __shfl_xor(v, 4, 32); v += __shfl_xor(v, 8, 32);
  return v;
}
static __device__ __forceinline__ float hmax32(float v) {
  v = hmax16(v); v = fmaxf(v, __shfl_xor(v, 16, 32)); return v;
}
static __device__ __forceinline__ float hsum32(float v) {
  v = hsum16(v); v += __shfl_xor(v, 16, 32); return v;
}

// ---- Y = [Res + relu](X[64 x Kin] @ W^T + bias) ----------------------------
// W: global f32 [Nout x Kin] row-major. Each wave owns n-blocks; one weight
// fragment is reused across all 4 row-blocks (4 independent accumulators).
// STORE_T: write result transposed (Y^T[Nout x 64], for the V projection).
template <bool STORE_T>
static __device__ __forceinline__ void gemm_xwT(
    const _Float16* Xs, int ldx, int Kin,
    GCF* Wg, GCF* bias, int Nout,
    _Float16* Ys, int ldy,
    const _Float16* Res, int ldr,
    int wave, int lane) {
  int nN = Nout >> 4;
  int n  = lane & 15;
  int rb = (lane & 16) ? 8 : 0;
  for (int nb = wave; nb < nN; nb += 4) {
    int n0 = nb << 4;
    v8f c[4] = {};
    for (int k0 = 0; k0 < Kin; k0 += 32) {
      v16h bf = loadBT_f32(Wg, Kin, n0, k0, lane);
      v16h a4[4];
#pragma unroll
      for (int m = 0; m < 4; ++m) a4[m] = loadA16(Xs, ldx, m << 4, k0, lane);
#pragma unroll
      for (int m = 0; m < 4; ++m)
        c[m] = __builtin_amdgcn_wmma_f32_16x16x32_f16(false, a4[m], false, bf,
                                                      (short)0, c[m], false, false);
    }
    float bn = bias[n0 + n];
#pragma unroll
    for (int m = 0; m < 4; ++m) {
#pragma unroll
      for (int r = 0; r < 8; ++r) {
        int row = (m << 4) + rb + r;
        float val = c[m][r] + bn;
        if (STORE_T) {
          Ys[(n0 + n) * ldy + row] = (_Float16)val;
        } else {
          if (Res) val = (float)Res[row * ldr + n0 + n] + fmaxf(val, 0.0f);
          Ys[row * ldy + n0 + n] = (_Float16)val;
        }
      }
    }
  }
}

// ---- One SAB stage with ShawRE; input X (64xDin f16 in sX), result -> sX ---
static __device__ void sab_stage(int Din,
    GCF* Wq, GCF* Wk, GCF* Wv, GCF* Wo,
    GCF* bq, GCF* bk, GCF* bv, GCF* bo,
    GCF* kp, GCF* vp, GCF* kpb,
    _Float16* sX, _Float16* sQ, _Float16* sK, _Float16* sVT, _Float16* sA,
    float* sQdot, float* sRowAK, int wave, int lane) {
  const float RS = 0.08838834764831845f; // 1/sqrt(128) (dimV = 128)
  gemm_xwT<false>(sX, LDH, Din, Wq, bq, 128, sQ, LDH, nullptr, 0, wave, lane);
  gemm_xwT<false>(sX, LDH, Din, Wk, bk, 128, sK, LDH, nullptr, 0, wave, lane);
  gemm_xwT<true >(sX, LDH, Din, Wv, bv, 128, sVT, LDT, nullptr, 0, wave, lane);
  __syncthreads();

  int n  = lane & 15;
  int rb = (lane & 16) ? 8 : 0;
  int m0 = wave * 16;          // wave owns 16 query rows

  for (int h = 0; h < 4; ++h) {
    // qdot[i] = Q_h[i,:] . kp   (per-row Shaw score bias)
    if (lane < 16) {
      int row = m0 + lane;
      float s = 0.f;
      for (int d = 0; d < 32; ++d)
        s += (float)sQ[row * LDH + 32 * h + d] * kp[d];
      sQdot[row] = s;
    }
    __syncthreads();

    // scores[m0:m0+16, 0:64] = Q_h @ K_h^T : 4 WMMA tiles, K=32
    v8f sc[4] = {};
    v16h aq = loadA16(sQ, LDH, m0, 32 * h, lane);
    v16h bk4[4];
#pragma unroll
    for (int t = 0; t < 4; ++t) bk4[t] = loadBT_f16(sK, LDH, t * 16, 32 * h, lane);
#pragma unroll
    for (int t = 0; t < 4; ++t)
      sc[t] = __builtin_amdgcn_wmma_f32_16x16x32_f16(false, aq, false, bk4[t],
                                                     (short)0, sc[t], false, false);

    // Shaw bias + softmax + rowAK, all in C-fragment registers.
    float kpv[4][8];
    float qd[8];
#pragma unroll
    for (int r = 0; r < 8; ++r) qd[r] = sQdot[m0 + rb + r];
#pragma unroll
    for (int t = 0; t < 4; ++t)
#pragma unroll
      for (int r = 0; r < 8; ++r)
        kpv[t][r] = kpb[(m0 + rb + r) * 64 + t * 16 + n];

#pragma unroll
    for (int r = 0; r < 8; ++r) {
      float m = -3.0e38f;
#pragma unroll
      for (int t = 0; t < 4; ++t) {
        float s = sc[t][r] * RS + qd[r] * kpv[t][r];
        sc[t][r] = s;
        m = fmaxf(m, s);
      }
      m = hmax16(m);                       // row lives in one 16-lane half
      float sum = 0.f, ak = 0.f;
#pragma unroll
      for (int t = 0; t < 4; ++t) {
        float e = __expf(sc[t][r] - m);
        sc[t][r] = e;
        sum += e;
        ak  += e * kpv[t][r];
      }
      sum = hsum16(sum);
      ak  = hsum16(ak);
      float inv = 1.f / sum;
#pragma unroll
      for (int t = 0; t < 4; ++t)
        sA[(m0 + rb + r) * LDA + t * 16 + n] = (_Float16)(sc[t][r] * inv);
      if (n == 0) sRowAK[m0 + rb + r] = ak * inv;
    }
    __syncthreads();

    // O_h = Q_h + A @ V_h + rowAK * vp  -> overwrite K_h columns (done with them)
    v16h aa[2];
#pragma unroll
    for (int kk = 0; kk < 2; ++kk) aa[kk] = loadA16(sA, LDA, m0, kk * 32, lane);
#pragma unroll
    for (int nt = 0; nt < 2; ++nt) {
      v16h bb[2];
#pragma unroll
      for (int kk = 0; kk < 2; ++kk)
        bb[kk] = loadBT_f16(sVT, LDT, 32 * h + nt * 16, kk * 32, lane);
      v8f c = {};
#pragma unroll
      for (int kk = 0; kk < 2; ++kk)
        c = __builtin_amdgcn_wmma_f32_16x16x32_f16(false, aa[kk], false, bb[kk],
                                                   (short)0, c, false, false);
      int col = 32 * h + nt * 16 + n;
      float vpn = vp[nt * 16 + n];
#pragma unroll
      for (int r = 0; r < 8; ++r) {
        int row = m0 + rb + r;
        float val = c[r] + (float)sQ[row * LDH + col] + sRowAK[row] * vpn;
        sK[row * LDH + col] = (_Float16)val;
      }
    }
    __syncthreads();
  }

  // X_next = O + relu(O @ Wo^T + bo)   (O lives in sK now)
  gemm_xwT<false>(sK, LDH, 128, Wo, bo, 128, sX, LDH, sK, LDH, wave, lane);
  __syncthreads();
}

struct WP {
  const float *s0Wq, *s0Wk, *s0Wv, *s0Wo, *s0bq, *s0bk, *s0bv, *s0bo, *s0kp, *s0vp;
  const float *s1Wq, *s1Wk, *s1Wv, *s1Wo, *s1bq, *s1bk, *s1bv, *s1bo, *s1kp, *s1vp;
  const float *dWq, *dWk, *dWv, *dWo, *dbq, *dbk, *dbv, *dbo;
  const float *Wfin, *bfin;
};

__global__ __launch_bounds__(128)
void whatsnet_kernel(const float* __restrict__ v, const float* __restrict__ q,
                     const float* __restrict__ kpos, WP P,
                     float* __restrict__ out) {
  __shared__ __align__(16) _Float16 sX[64 * LDH];
  __shared__ __align__(16) _Float16 sQ[64 * LDH];
  __shared__ __align__(16) _Float16 sK[64 * LDH];
  __shared__ __align__(16) _Float16 sVT[128 * LDT];   // V stored transposed
  __shared__ __align__(16) _Float16 sA[64 * LDA];
  __shared__ float sQdot[64];
  __shared__ float sRowAK[64];
  __shared__ float sVecQ[128];
  __shared__ float sVecA[256];
  __shared__ float sVecO[128];
  __shared__ float sVecF[128];

  const int b    = blockIdx.x;
  const int tid  = threadIdx.x;
  const int wave = tid >> 5;
  const int lane = tid & 31;
  GCF* vb  = gptr(v)    + (size_t)b * 4096;
  GCF* qb  = gptr(q)    + (size_t)b * 64;
  GCF* kpb = gptr(kpos) + (size_t)b * 4096;
  GMF* outp = gptr_mut(out);

  for (int i = tid; i < 4096; i += 128)
    sX[(i >> 6) * LDH + (i & 63)] = (_Float16)vb[i];
  __syncthreads();

  sab_stage(64, gptr(P.s0Wq), gptr(P.s0Wk), gptr(P.s0Wv), gptr(P.s0Wo),
            gptr(P.s0bq), gptr(P.s0bk), gptr(P.s0bv), gptr(P.s0bo),
            gptr(P.s0kp), gptr(P.s0vp), kpb,
            sX, sQ, sK, sVT, sA, sQdot, sRowAK, wave, lane);
  sab_stage(128, gptr(P.s1Wq), gptr(P.s1Wk), gptr(P.s1Wv), gptr(P.s1Wo),
            gptr(P.s1bq), gptr(P.s1bk), gptr(P.s1bv), gptr(P.s1bo),
            gptr(P.s1kp), gptr(P.s1vp), kpb,
            sX, sQ, sK, sVT, sA, sQdot, sRowAK, wave, lane);

  // ---- decoder MAB (Q = q_b [1x64], K = h in sX) ----
  gemm_xwT<false>(sX, LDH, 128, gptr(P.dWk), gptr(P.dbk), 128, sK, LDH,
                  nullptr, 0, wave, lane);
  gemm_xwT<true >(sX, LDH, 128, gptr(P.dWv), gptr(P.dbv), 128, sVT, LDT,
                  nullptr, 0, wave, lane);
  __syncthreads();

  {  // Qp_dec = q @ Wq^T + bq   (1x128, VALU)
    float s = gptr(P.dbq)[tid];
    GCF* wr = gptr(P.dWq) + (size_t)tid * 64;
    for (int d = 0; d < 64; ++d) s += qb[d] * wr[d];
    sVecQ[tid] = s;
  }
  __syncthreads();

  {  // per-head single-row attention: wave == head
    const float RS = 0.08838834764831845f;
    int h = wave;
    float s1 = 0.f, s2 = 0.f;
    for (int d = 0; d < 32; ++d) {
      float qd = sVecQ[32 * h + d];
      s1 += qd * (float)sK[lane * LDH + 32 * h + d];
      s2 += qd * (float)sK[(lane + 32) * LDH + 32 * h + d];
    }
    s1 *= RS; s2 *= RS;
    float m = hmax32(fmaxf(s1, s2));
    float e1 = __expf(s1 - m), e2 = __expf(s2 - m);
    float ssum = hsum32(e1 + e2);
    float inv = 1.f / ssum;
    sVecA[h * 64 + lane]      = e1 * inv;
    sVecA[h * 64 + lane + 32] = e2 * inv;
  }
  __syncthreads();

  {  // O = Q_ + A @ V_   (V transposed: per-lane contiguous j)
    int h = wave;
    float acc = sVecQ[32 * h + lane];
    const _Float16* vrow = sVT + (32 * h + lane) * LDT;
    for (int j = 0; j < 64; ++j)
      acc += sVecA[h * 64 + j] * (float)vrow[j];
    sVecO[32 * h + lane] = acc;
  }
  __syncthreads();

  {  // O + relu(O @ Wo^T + bo)
    float s = gptr(P.dbo)[tid];
    GCF* wr = gptr(P.dWo) + (size_t)tid * 128;
    for (int d = 0; d < 128; ++d) s += sVecO[d] * wr[d];
    sVecF[tid] = sVecO[tid] + fmaxf(s, 0.f);
  }
  __syncthreads();

  if (tid < 64) {  // final linear -> out[b, 0, :64]
    float s = gptr(P.bfin)[tid];
    GCF* wr = gptr(P.Wfin) + (size_t)tid * 128;
    for (int d = 0; d < 128; ++d) s += sVecF[d] * wr[d];
    outp[(size_t)b * 64 + tid] = s;
  }
}

extern "C" void kernel_launch(void* const* d_in, const int* in_sizes, int n_in,
                              void* d_out, int out_size, void* d_ws, size_t ws_size,
                              hipStream_t stream) {
  // Input order = JAX pytree flatten (dict keys sorted alphabetically):
  // 0:v 1:q 2:kpos 3:Wfin 4:bfin
  // 5..12:  dec  {Wk,Wo,Wq,Wv,bk,bo,bq,bv}
  // 13..22: sab0 {Wk,Wo,Wq,Wv,bk,bo,bq,bv,kp,vp}
  // 23..32: sab1 {Wk,Wo,Wq,Wv,bk,bo,bq,bv,kp,vp}
  const float* v    = (const float*)d_in[0];
  const float* q    = (const float*)d_in[1];
  const float* kpos = (const float*)d_in[2];
  WP P;
  P.Wfin = (const float*)d_in[3];  P.bfin = (const float*)d_in[4];
  P.dWk  = (const float*)d_in[5];  P.dWo  = (const float*)d_in[6];
  P.dWq  = (const float*)d_in[7];  P.dWv  = (const float*)d_in[8];
  P.dbk  = (const float*)d_in[9];  P.dbo  = (const float*)d_in[10];
  P.dbq  = (const float*)d_in[11]; P.dbv  = (const float*)d_in[12];
  P.s0Wk = (const float*)d_in[13]; P.s0Wo = (const float*)d_in[14];
  P.s0Wq = (const float*)d_in[15]; P.s0Wv = (const float*)d_in[16];
  P.s0bk = (const float*)d_in[17]; P.s0bo = (const float*)d_in[18];
  P.s0bq = (const float*)d_in[19]; P.s0bv = (const float*)d_in[20];
  P.s0kp = (const float*)d_in[21]; P.s0vp = (const float*)d_in[22];
  P.s1Wk = (const float*)d_in[23]; P.s1Wo = (const float*)d_in[24];
  P.s1Wq = (const float*)d_in[25]; P.s1Wv = (const float*)d_in[26];
  P.s1bk = (const float*)d_in[27]; P.s1bo = (const float*)d_in[28];
  P.s1bq = (const float*)d_in[29]; P.s1bv = (const float*)d_in[30];
  P.s1kp = (const float*)d_in[31]; P.s1vp = (const float*)d_in[32];

  int B = in_sizes[0] / 4096;  // v is [B, 64, 64]
  whatsnet_kernel<<<B, 128, 0, stream>>>(v, q, kpos, P, (float*)d_out);
  (void)n_in; (void)out_size; (void)d_ws; (void)ws_size;
}
